// Edge_conv_19971597926624
// MI455X (gfx1250) — matrix operand order repaired and checked by reading, not compiled, not run on hardware
//
#include <hip/hip_runtime.h>

// EdgeConv (B=2, N=16384, C=256, INTER=128, K=16) for gfx1250 (MI455X).
// Pipeline: kNN (fp32, LDS-tiled) -> pack W1/W2 into WMMA B-fragment layout
// -> fused gather + 2x GEMM via v_wmma_f32_16x16x32_f16 + softmax reduce.

#define BATCH 2
#define NPTS  16384
#define CFEAT 256
#define INTER 128
#define KNN   16

typedef _Float16 v16h __attribute__((ext_vector_type(16)));
typedef _Float16 v8h  __attribute__((ext_vector_type(8)));
typedef float    v8f  __attribute__((ext_vector_type(8)));

// ---------------------------------------------------------------------------
// Kernel 1: brute-force 16-NN per point (squared distances), one thread/query.
// ---------------------------------------------------------------------------
__launch_bounds__(256)
__global__ void knn_kernel(const float* __restrict__ coords,
                           float* __restrict__ knn_d,
                           int*   __restrict__ knn_i) {
  __shared__ float sc[512 * 3];
  const int tid = threadIdx.x;
  const int q   = blockIdx.x * 256 + tid;      // global query id
  const int b   = q >> 14;                     // N = 16384
  const int n   = q & (NPTS - 1);
  const float* cb = coords + (size_t)b * NPTS * 3;

  const float qx = cb[n * 3 + 0];
  const float qy = cb[n * 3 + 1];
  const float qz = cb[n * 3 + 2];

  float bd[KNN];
  int   bi[KNN];
#pragma unroll
  for (int i = 0; i < KNN; ++i) { bd[i] = 3.4e38f; bi[i] = 0; }
  float wmax = 3.4e38f;
  int   wslot = 0;

  for (int t = 0; t < NPTS / 512; ++t) {
    __syncthreads();
    for (int i = tid; i < 1536; i += 256) sc[i] = cb[t * 1536 + i];
    __syncthreads();
    for (int p = 0; p < 512; ++p) {
      const float dx = sc[p * 3 + 0] - qx;
      const float dy = sc[p * 3 + 1] - qy;
      const float dz = sc[p * 3 + 2] - qz;
      const float dd = dx * dx + dy * dy + dz * dz;
      if (dd < wmax) {
        const int gi = t * 512 + p;
#pragma unroll
        for (int i = 0; i < KNN; ++i)
          if (i == wslot) { bd[i] = dd; bi[i] = gi; }
        wmax = -1.0f;
#pragma unroll
        for (int i = 0; i < KNN; ++i)
          if (bd[i] > wmax) { wmax = bd[i]; wslot = i; }
      }
    }
  }
#pragma unroll
  for (int i = 0; i < KNN; ++i) {
    knn_d[q * KNN + i] = bd[i];
    knn_i[q * KNN + i] = bi[i];
  }
}

// ---------------------------------------------------------------------------
// Kernels 2/3: pack W1 (512x128) and W2 (128x256) into WMMA B-fragment order.
// Fragment (kt,nt): lane n holds column nt*16 + n%16, 16 contiguous K halves
// (K = kt*32 + (n<16 ? 0 : 16) + j).
// ---------------------------------------------------------------------------
__global__ void pack_w1_kernel(const float* __restrict__ W1,
                               _Float16* __restrict__ w1h) {
  const int t    = blockIdx.x * blockDim.x + threadIdx.x;   // 0..65535
  const int j    = t & 15;
  const int lane = (t >> 4) & 31;
  const int nt   = (t >> 9) & 7;
  const int kt   = t >> 12;                                 // 0..15
  const int k    = kt * 32 + ((lane & 16) ? 16 : 0) + j;
  const int col  = nt * 16 + (lane & 15);
  w1h[t] = (_Float16)W1[k * INTER + col];
}

__global__ void pack_w2_kernel(const float* __restrict__ W2,
                               _Float16* __restrict__ w2h) {
  const int t    = blockIdx.x * blockDim.x + threadIdx.x;   // 0..32767
  const int j    = t & 15;
  const int lane = (t >> 4) & 31;
  const int nt   = (t >> 9) & 15;
  const int kt   = t >> 13;                                 // 0..3
  const int k    = kt * 32 + ((lane & 16) ? 16 : 0) + j;
  const int col  = nt * 16 + (lane & 15);
  w2h[t] = (_Float16)W2[k * CFEAT + col];
}

// ---------------------------------------------------------------------------
// Kernel 4: one wave per query. grouped(16x512) @ W1 -> relu -> @ W2 ->
// softmax(-d)-weighted row sum -> out(256). All matmuls on v_wmma.
// ---------------------------------------------------------------------------
__device__ __forceinline__ float4 f4sub(float4 a, float4 b) {
  return make_float4(a.x - b.x, a.y - b.y, a.z - b.z, a.w - b.w);
}

__device__ __forceinline__ void cvt8(v16h& a, int off, float4 x0, float4 x1) {
  a[off + 0] = (_Float16)x0.x; a[off + 1] = (_Float16)x0.y;
  a[off + 2] = (_Float16)x0.z; a[off + 3] = (_Float16)x0.w;
  a[off + 4] = (_Float16)x1.x; a[off + 5] = (_Float16)x1.y;
  a[off + 6] = (_Float16)x1.z; a[off + 7] = (_Float16)x1.w;
}

__launch_bounds__(256)
__global__ void edge_kernel(const float* __restrict__ feats,
                            const float* __restrict__ b1,
                            const float* __restrict__ b2,
                            const _Float16* __restrict__ w1h,
                            const _Float16* __restrict__ w2h,
                            const float* __restrict__ knn_d,
                            const int*   __restrict__ knn_i,
                            float* __restrict__ out) {
  // padded (136-half row stride) to avoid LDS bank conflicts on the transpose
  __shared__ __align__(16) _Float16 hbuf[8][16][136];

  const int tid   = threadIdx.x;
  const int wv    = tid >> 5;
  const int lane  = tid & 31;
  const int q     = blockIdx.x * 8 + wv;       // query id (0..32767)
  const int b     = q >> 14;
  const int n     = q & (NPTS - 1);
  const int col16 = lane & 15;
  const int hi    = lane >> 4;                 // 0 or 1 (half-wave)
  const int rowoff = hi * 8;

  // ---- softmax(-d) over the 16 neighbors (intra-16-lane reductions) ----
  const float d = knn_d[q * KNN + col16];
  float m = d;
  m = fminf(m, __shfl_xor(m, 1, 32));
  m = fminf(m, __shfl_xor(m, 2, 32));
  m = fminf(m, __shfl_xor(m, 4, 32));
  m = fminf(m, __shfl_xor(m, 8, 32));
  const float e = __expf(m - d);               // exp(-d - max(-d))
  float s = e;
  s += __shfl_xor(s, 1, 32);
  s += __shfl_xor(s, 2, 32);
  s += __shfl_xor(s, 4, 32);
  s += __shfl_xor(s, 8, 32);
  const float w = e / s;
  float wrow[8];
#pragma unroll
  for (int v = 0; v < 8; ++v) wrow[v] = __shfl(w, rowoff + v, 32);

  // this lane's A-matrix row = neighbor col16
  const int g = knn_i[q * KNN + col16];
  const float* fg = feats + ((size_t)b * NPTS + g) * CFEAT;
  const float* fc = feats + ((size_t)b * NPTS + n) * CFEAT;

  // ---- GEMM1: grouped(16x512) @ W1(512x128), K in 16 chunks of 32 ----
  const v8f zf = {0.f, 0.f, 0.f, 0.f, 0.f, 0.f, 0.f, 0.f};
  v8f acc1[8];
#pragma unroll
  for (int nt = 0; nt < 8; ++nt) acc1[nt] = zf;

  const v16h* w1f = (const v16h*)w1h;
#pragma unroll
  for (int c = 0; c < 16; ++c) {
    const int base = c * 32 + hi * 16;         // start of this lane's 16 contiguous K
    v16h a;
    if (c < 8) {                               // nn_feats half of grouped
      const float4 x0 = *(const float4*)(fg + base);
      const float4 x1 = *(const float4*)(fg + base + 4);
      const float4 y0 = *(const float4*)(fg + base + 8);
      const float4 y1 = *(const float4*)(fg + base + 12);
      cvt8(a, 0, x0, x1);
      cvt8(a, 8, y0, y1);
    } else {                                   // delta = nn_feats - center
      const int fb = base - 256;
      const float4 x0 = f4sub(*(const float4*)(fg + fb),
                              *(const float4*)(fc + fb));
      const float4 x1 = f4sub(*(const float4*)(fg + fb + 4),
                              *(const float4*)(fc + fb + 4));
      const float4 y0 = f4sub(*(const float4*)(fg + fb + 8),
                              *(const float4*)(fc + fb + 8));
      const float4 y1 = f4sub(*(const float4*)(fg + fb + 12),
                              *(const float4*)(fc + fb + 12));
      cvt8(a, 0, x0, x1);
      cvt8(a, 8, y0, y1);
    }
#pragma unroll
    for (int nt = 0; nt < 8; ++nt) {
      const v16h bf = w1f[(c * 8 + nt) * 32 + lane];
      acc1[nt] = __builtin_amdgcn_wmma_f32_16x16x32_f16(
          false, a, false, bf, (short)0, acc1[nt], false, false);
    }
  }

  // ---- bias + relu, spill h (16x128) to LDS as f16 (C-layout -> rows) ----
#pragma unroll
  for (int nt = 0; nt < 8; ++nt) {
    const float bias = b1[nt * 16 + col16];
#pragma unroll
    for (int v = 0; v < 8; ++v) {
      float hv = acc1[nt][v] + bias;
      hv = fmaxf(hv, 0.0f);
      hbuf[wv][rowoff + v][nt * 16 + col16] = (_Float16)hv;
    }
  }
  asm volatile("s_wait_dscnt 0" ::: "memory");

  // ---- reload h as A-fragments for GEMM2 (4 K-chunks of 32) ----
  v16h a2[4];
#pragma unroll
  for (int kt = 0; kt < 4; ++kt) {
    const int k0 = kt * 32 + hi * 16;
    const v8h lo  = *(const v8h*)&hbuf[wv][col16][k0];
    const v8h hi8 = *(const v8h*)&hbuf[wv][col16][k0 + 8];
#pragma unroll
    for (int j = 0; j < 8; ++j) { a2[kt][j] = lo[j]; a2[kt][8 + j] = hi8[j]; }
  }

  // ---- GEMM2 (h @ W2) fused with softmax-weighted row reduction ----
  const v16h* w2f = (const v16h*)w2h;
#pragma unroll
  for (int nt = 0; nt < 16; ++nt) {
    v8f acc = zf;
#pragma unroll
    for (int kt = 0; kt < 4; ++kt) {
      const v16h bf = w2f[(kt * 16 + nt) * 32 + lane];
      acc = __builtin_amdgcn_wmma_f32_16x16x32_f16(
          false, a2[kt], false, bf, (short)0, acc, false, false);
    }
    // lane holds column nt*16+col16; VGPR v = row rowoff+v
    float partial = 0.0f;
#pragma unroll
    for (int v = 0; v < 8; ++v) partial += wrow[v] * acc[v];
    partial += __shfl_xor(partial, 16, 32);    // combine both row halves
    if (lane < 16)
      out[(size_t)q * CFEAT + nt * 16 + lane] = partial + b2[nt * 16 + lane];
  }
}

// ---------------------------------------------------------------------------
extern "C" void kernel_launch(void* const* d_in, const int* in_sizes, int n_in,
                              void* d_out, int out_size, void* d_ws, size_t ws_size,
                              hipStream_t stream) {
  const float* coords = (const float*)d_in[0];
  const float* feats  = (const float*)d_in[1];
  const float* W1     = (const float*)d_in[2];
  const float* b1     = (const float*)d_in[3];
  const float* W2     = (const float*)d_in[4];
  const float* b2     = (const float*)d_in[5];
  float* out = (float*)d_out;

  // workspace carve-up (all 16B-aligned): 2MB dists, 2MB idx, packed weights
  char* ws = (char*)d_ws;
  float*    knn_d = (float*)ws;                                      // 2 MB
  int*      knn_i = (int*)(ws + (size_t)2 * 1024 * 1024);            // 2 MB
  _Float16* w1h   = (_Float16*)(ws + (size_t)4 * 1024 * 1024);       // 128 KB
  _Float16* w2h   = (_Float16*)(ws + (size_t)4 * 1024 * 1024 + 131072); // 64 KB

  knn_kernel<<<BATCH * NPTS / 256, 256, 0, stream>>>(coords, knn_d, knn_i);
  pack_w1_kernel<<<(2 * CFEAT * INTER) / 256, 256, 0, stream>>>(W1, w1h);
  pack_w2_kernel<<<(INTER * CFEAT) / 256, 256, 0, stream>>>(W2, w2h);
  edge_kernel<<<BATCH * NPTS / 8, 256, 0, stream>>>(feats, b1, b2, w1h, w2h,
                                                    knn_d, knn_i, out);
}
